// MolConv_7275674599511
// MI455X (gfx1250) — compile-verified, matrix operand order checked by reading, not run on hardware
//
#include <hip/hip_runtime.h>
#include <hip/hip_bf16.h>

#define BB 16
#define CC 32
#define NN 2048
#define KK 24
#define BNKTOT (BB * NN * KK) // 786432

typedef __attribute__((ext_vector_type(16))) _Float16 v16h;
typedef __attribute__((ext_vector_type(8)))  _Float16 v8h;
typedef __attribute__((ext_vector_type(8)))  float    v8f;
typedef __attribute__((ext_vector_type(4)))  float    v4f;

// Load a 16x32 f16 WMMA A/B fragment from LDS.
// base points at element [row0][0], rows are stride_h halfwords apart (16B aligned).
// Layout per ISA: lanes 0-15 hold row (rc0+lane), halves 0..7 = K 0..7, halves 8..15 = K 16..23;
// lanes 16-31 hold the same rows with K 8..15 / 24..31.
__device__ __forceinline__ v16h ld_frag(const _Float16* base, int stride_h, int rc0, int lane) {
  const _Float16* p = base + (rc0 + (lane & 15)) * stride_h + ((lane >> 4) << 3);
  v8h lo = *(const v8h*)(p);
  v8h hi = *(const v8h*)(p + 16);
  v16h f;
#pragma unroll
  for (int i = 0; i < 8; ++i) { f[i] = lo[i]; f[i + 8] = hi[i]; }
  return f;
}

__device__ __forceinline__ v8f wmma16(v16h a, v16h b, v8f c) {
  return __builtin_amdgcn_wmma_f32_16x16x32_f16(false, a, false, b, (short)0, c, false, false);
}

// ---------------------------------------------------------------------------
// Kernel 1: per-batch gram matrix tiles (WMMA) + top-K selection per row.
// grid (N/16, B), 256 threads (8 waves). Dynamic LDS: 270336 B.
// ---------------------------------------------------------------------------
__global__ void topk_kernel(const float* __restrict__ x,
                            int* __restrict__ idx_ws,
                            float* __restrict__ dist_ws) {
  extern __shared__ char smem[];
  _Float16* xh  = (_Float16*)smem;            // [2048][32] f16 -> 131072 B
  float* xxs    = (float*)(smem + 131072);    // [2048]          ->   8192 B
  float* pairb  = (float*)(smem + 139264);    // [16][2048]      -> 131072 B

  const int tid = threadIdx.x;
  const int wave = tid >> 5, lane = tid & 31;
  const int b = blockIdx.y;
  const int n0 = blockIdx.x * 16;
  const float* xb = x + (size_t)b * CC * NN;

  // stage x[b] transposed into LDS as f16 [n][c]
  for (int e = tid; e < CC * NN; e += 256) {
    int c = e >> 11, n = e & (NN - 1);
    xh[n * CC + c] = (_Float16)xb[e];
  }
  __syncthreads();
  // squared norms (from the same f16 values so self-distance ~ 0 stays maximal)
  for (int n = tid; n < NN; n += 256) {
    float s = 0.f;
#pragma unroll
    for (int c = 0; c < CC; ++c) { float v = (float)xh[n * CC + c]; s += v * v; }
    xxs[n] = s;
  }
  __syncthreads();

  // pair(n, m) = 2*dot - xx[n] - xx[m]; one WMMA per 16x16 tile (K = C = 32)
  v16h afrag = ld_frag(xh, CC, n0, lane);
  const int lro = (lane >> 4) << 3;
  float xxrow[8];                     // hoisted: row norms live in registers
#pragma unroll
  for (int r = 0; r < 8; ++r) xxrow[r] = xxs[n0 + r + lro];
  for (int j = 0; j < 16; ++j) {
    int m0 = (wave + j * 8) * 16;
    v16h bfrag = ld_frag(xh, CC, m0, lane);
    v8f acc = {};
    acc = wmma16(afrag, bfrag, acc);
    int m = m0 + (lane & 15);
    float xxm = xxs[m];
#pragma unroll
    for (int r = 0; r < 8; ++r) {
      int lr = r + lro;
      pairb[lr * NN + m] = 2.f * acc[r] - xxrow[r] - xxm;
    }
  }
  __syncthreads();

  // iterative argmax top-K per row, 2 rows per wave; 4-wide LDS scans
  for (int li = 0; li < 2; ++li) {
    int lr = wave * 2 + li;
    float* prow = pairb + lr * NN;
    const v4f* prow4 = (const v4f*)prow;
    size_t ob = ((size_t)(b * NN) + n0 + lr) * KK;
    for (int k = 0; k < KK; ++k) {
      float best = -3.0e38f; int bidx = 0;
      for (int jj = 0; jj < NN / 128; ++jj) {   // 16 x ds_load_b128 per lane
        int vi = jj * 32 + lane;
        v4f v = prow4[vi];
        int mbase = vi * 4;
#pragma unroll
        for (int ii = 0; ii < 4; ++ii) {
          if (v[ii] > best) { best = v[ii]; bidx = mbase + ii; }
        }
      }
#pragma unroll
      for (int off = 16; off >= 1; off >>= 1) {
        float ov = __shfl_xor(best, off);
        int   oi = __shfl_xor(bidx, off);
        if (ov > best || (ov == best && oi < bidx)) { best = ov; bidx = oi; }
      }
      if (lane == 0) {
        idx_ws[ob + k]  = bidx;
        dist_ws[ob + k] = sqrtf(fmaxf(-best, 1e-12f));
        prow[bidx] = -3.0e38f; // remove winner; DS ops are in-order within a wave
      }
      asm volatile("" ::: "memory"); // pin compiler ordering around the kill-store
      __builtin_amdgcn_wave_barrier();
    }
  }
}

__global__ void zero_stats_kernel(float* __restrict__ stats) {
  if (threadIdx.x < 128) stats[threadIdx.x] = 0.f;
}

// ---------------------------------------------------------------------------
// Kernel 2: attention MLP (WMMA, K padded 80->96) + softmax + BN statistics.
// grid (N/16, B), 256 threads. Dynamic LDS: 119296 B.
// ---------------------------------------------------------------------------
__global__ void attn_kernel(const float* __restrict__ x,
                            const float* __restrict__ w1,
                            const float* __restrict__ w2,
                            const float* __restrict__ updw,
                            const int* __restrict__ idx_ws,
                            const float* __restrict__ dist_ws,
                            float* __restrict__ att_ws,
                            float* __restrict__ stats_g) {
  extern __shared__ char smem[];
  _Float16* attin = (_Float16*)smem;            // [384][96]
  _Float16* w1h   = attin + 384 * 96;           // [64][96]
  _Float16* updh  = w1h + 64 * 96;              // [64][32]
  _Float16* wbuf  = updh + 64 * 32;             // [384][32]
  float* logitsL  = (float*)(wbuf + 384 * 32);  // [384]
  float* attL     = logitsL + 384;              // [384]
  float* statsL   = attL + 384;                 // [128]
  _Float16* centh = (_Float16*)(statsL + 128);  // [16][32] staged center cols

  const int tid = threadIdx.x;
  const int wave = tid >> 5, lane = tid & 31;
  const int b = blockIdx.y;
  const int n0 = blockIdx.x * 16;
  const float* xb = x + (size_t)b * CC * NN;

  for (int r = tid; r < 384; r += 256) logitsL[r] = 0.f;
  if (tid < 128) statsL[tid] = 0.f;
  for (int e = tid; e < 64 * 96; e += 256) {
    int o = e / 96, f = e - o * 96;
    w1h[e] = (f < 80) ? (_Float16)w1[o * 80 + f] : (_Float16)0.f;
  }
  for (int e = tid; e < 64 * 32; e += 256) updh[e] = (_Float16)updw[e];
  // stage the 16 center columns once (coalesced: consecutive n per c)
  for (int e = tid; e < 16 * 32; e += 256) {
    int c = e >> 4, i = e & 15;
    centh[i * CC + c] = (_Float16)xb[c * NN + n0 + i];
  }
  __syncthreads();

  // att_in rows: [cent(32) | neigh(32) | rbf(16) | zero-pad(16)]
  for (int r = tid; r < 384; r += 256) {
    int i = r / 24, k = r - i * 24;
    size_t pb = ((size_t)(b * NN) + n0 + i) * KK + k;
    int m = idx_ws[pb];
    float d = dist_ws[pb];
    _Float16* row = attin + r * 96;
    const v8h* csrc = (const v8h*)(centh + i * CC);
#pragma unroll
    for (int c8 = 0; c8 < 4; ++c8) ((v8h*)row)[c8] = csrc[c8]; // b128 LDS copies
#pragma unroll
    for (int c = 0; c < CC; ++c) row[CC + c] = (_Float16)xb[c * NN + m];
#pragma unroll
    for (int j = 0; j < 16; ++j) {
      float diff = d - (float)j * (5.0f / 15.0f);
      float v = expf(-10.0f * diff * diff);
      v = fminf(fmaxf(v, 1e-10f), 1.0f);
      row[64 + j] = (_Float16)v;
    }
  }
  __syncthreads();

  // GEMM1 (384x96 @ 96x64) fused with leaky(0.2) and the 64->1 projection
  for (int t = wave; t < 24; t += 8) {
    v16h a0 = ld_frag(attin,      96, t * 16, lane);
    v16h a1 = ld_frag(attin + 32, 96, t * 16, lane);
    v16h a2 = ld_frag(attin + 64, 96, t * 16, lane);
    for (int ot = 0; ot < 4; ++ot) {
      v16h b0 = ld_frag(w1h,      96, ot * 16, lane);
      v16h b1 = ld_frag(w1h + 32, 96, ot * 16, lane);
      v16h b2 = ld_frag(w1h + 64, 96, ot * 16, lane);
      v8f acc = {};
      acc = wmma16(a0, b0, acc);
      acc = wmma16(a1, b1, acc);
      acc = wmma16(a2, b2, acc);
      float w2v = w2[ot * 16 + (lane & 15)];
#pragma unroll
      for (int r = 0; r < 8; ++r) {
        float h = acc[r];
        h = (h > 0.f) ? h : 0.2f * h;
        float s = h * w2v;
        s += __shfl_xor(s, 1); s += __shfl_xor(s, 2);
        s += __shfl_xor(s, 4); s += __shfl_xor(s, 8);
        if ((lane & 15) == 0) {
          int rrow = t * 16 + r + ((lane >> 4) << 3);
          logitsL[rrow] += s; // rows are wave-exclusive
        }
      }
    }
  }
  __syncthreads();

  // softmax over K (2 points per wave, lanes 0..23 active)
  for (int ii = 0; ii < 2; ++ii) {
    int i = wave * 2 + ii;
    float v = (lane < KK) ? logitsL[i * KK + lane] : -3.0e38f;
    float mx = v;
#pragma unroll
    for (int off = 16; off >= 1; off >>= 1) mx = fmaxf(mx, __shfl_xor(mx, off));
    float e = (lane < KK) ? expf(v - mx) : 0.f;
    float sm = e;
#pragma unroll
    for (int off = 16; off >= 1; off >>= 1) sm += __shfl_xor(sm, off);
    float a = e / sm;
    if (lane < KK) {
      attL[i * KK + lane] = a;
      att_ws[((size_t)(b * NN) + n0 + i) * KK + lane] = a;
    }
  }
  __syncthreads();

  for (int r = tid; r < 384; r += 256) {
    float a = attL[r];
    const _Float16* nrow = attin + r * 96 + 32;
#pragma unroll
    for (int c = 0; c < CC; ++c) wbuf[r * CC + c] = (_Float16)(a * (float)nrow[c]);
  }
  __syncthreads();

  // GEMM2 (384x32 @ 32x64) just for BN sum/sumsq accumulation
  float s1[4] = {0.f, 0.f, 0.f, 0.f};
  float s2[4] = {0.f, 0.f, 0.f, 0.f};
  for (int t = wave; t < 24; t += 8) {
    v16h a = ld_frag(wbuf, 32, t * 16, lane);
    for (int ot = 0; ot < 4; ++ot) {
      v16h bf = ld_frag(updh, 32, ot * 16, lane);
      v8f acc = {};
      acc = wmma16(a, bf, acc);
#pragma unroll
      for (int r = 0; r < 8; ++r) { float u = acc[r]; s1[ot] += u; s2[ot] += u * u; }
    }
  }
#pragma unroll
  for (int ot = 0; ot < 4; ++ot) {
    int o = ot * 16 + (lane & 15);
    atomicAdd(&statsL[o], s1[ot]);
    atomicAdd(&statsL[64 + o], s2[ot]);
  }
  __syncthreads();
  if (tid < 128) atomicAdd(&stats_g[tid], statsL[tid]);
}

// ---------------------------------------------------------------------------
// Kernel 3: BN affine + leaky(0.02) + mean over K + residual path + transpose.
// grid (N/16, B), 256 threads. Static LDS ~38 KB.
// ---------------------------------------------------------------------------
__global__ void finalize_kernel(const float* __restrict__ x,
                                const float* __restrict__ updw,
                                const float* __restrict__ resw,
                                const float* __restrict__ gamma,
                                const float* __restrict__ beta,
                                const int* __restrict__ idx_ws,
                                const float* __restrict__ att_ws,
                                const float* __restrict__ stats_g,
                                float* __restrict__ out) {
  __shared__ __attribute__((aligned(16))) _Float16 wbuf[384 * 32];
  __shared__ __attribute__((aligned(16))) _Float16 updh[64 * 32];
  __shared__ __attribute__((aligned(16))) _Float16 reswh[64 * 32];
  __shared__ __attribute__((aligned(16))) _Float16 meanNh[16 * 32];
  __shared__ float featL[16 * 64];
  __shared__ float scaleL[64];
  __shared__ float shiftL[64];

  const int tid = threadIdx.x;
  const int wave = tid >> 5, lane = tid & 31;
  const int b = blockIdx.y;
  const int n0 = blockIdx.x * 16;
  const float* xb = x + (size_t)b * CC * NN;

  if (tid < 64) {
    const float inv = 1.0f / (float)BNKTOT;
    float mean = stats_g[tid] * inv;
    float var = stats_g[64 + tid] * inv - mean * mean;
    float sc = gamma[tid] * rsqrtf(var + 1e-5f);
    scaleL[tid] = sc;
    shiftL[tid] = beta[tid] - mean * sc;
  }
  for (int e = tid; e < 16 * 64; e += 256) featL[e] = 0.f;
  for (int e = tid; e < 64 * 32; e += 256) {
    updh[e]  = (_Float16)updw[e];
    reswh[e] = (_Float16)resw[e];
  }
  for (int r = tid; r < 384; r += 256) {
    int i = r / 24, k = r - i * 24;
    size_t pb = ((size_t)(b * NN) + n0 + i) * KK + k;
    int m = idx_ws[pb];
    float a = att_ws[pb];
#pragma unroll
    for (int c = 0; c < CC; ++c) wbuf[r * CC + c] = (_Float16)(a * xb[c * NN + m]);
  }
  for (int e = tid; e < 16 * 32; e += 256) {
    int i = e >> 5, c = e & 31;
    size_t pb = ((size_t)(b * NN) + n0 + i) * KK;
    float s = 0.f;
    for (int k = 0; k < KK; ++k) s += xb[c * NN + idx_ws[pb + k]];
    meanNh[e] = (_Float16)(s * (1.0f / 24.0f));
  }
  __syncthreads();

  // upd GEMM: wave w owns rows [48w, 48w+48) == points 2w, 2w+1 exclusively
  const int i0 = wave * 2, i1 = wave * 2 + 1;
  const int rowsplit = wave * 48 + 24;
  for (int ot = 0; ot < 4; ++ot) {
    int o = ot * 16 + (lane & 15);
    float sc = scaleL[o], sh = shiftL[o];
    v16h bf = ld_frag(updh, 32, ot * 16, lane);
    float f0 = 0.f, f1 = 0.f;
    for (int tt = 0; tt < 3; ++tt) {
      int t = wave * 3 + tt;
      v16h a = ld_frag(wbuf, 32, t * 16, lane);
      v8f acc = {};
      acc = wmma16(a, bf, acc);
#pragma unroll
      for (int r = 0; r < 8; ++r) {
        int row = t * 16 + r + ((lane >> 4) << 3);
        float v = acc[r] * sc + sh;
        v = (v > 0.f) ? v : 0.02f * v;
        if (row < rowsplit) f0 += v; else f1 += v;
      }
    }
    f0 += __shfl_xor(f0, 16);
    f1 += __shfl_xor(f1, 16);
    if (lane < 16) { // o distinct per lane, i exclusive per wave -> no atomics
      featL[i0 * 64 + o] += f0;
      featL[i1 * 64 + o] += f1;
    }
  }
  __syncthreads();

  // residual: (mean_k neigh) @ res_w^T, one o-tile per wave (waves 0..3)
  if (wave < 4) {
    int ot = wave;
    v16h a  = ld_frag(meanNh, 32, 0, lane);
    v16h bf = ld_frag(reswh, 32, ot * 16, lane);
    v8f acc = {};
    acc = wmma16(a, bf, acc);
#pragma unroll
    for (int r = 0; r < 8; ++r) {
      int i = r + ((lane >> 4) << 3);
      int o = ot * 16 + (lane & 15);
      float val = featL[i * 64 + o] * (1.0f / 24.0f) + 0.1f * acc[r];
      out[((size_t)b * 64 + o) * NN + n0 + i] = val;
    }
  }
}

// ---------------------------------------------------------------------------
extern "C" void kernel_launch(void* const* d_in, const int* in_sizes, int n_in,
                              void* d_out, int out_size, void* d_ws, size_t ws_size,
                              hipStream_t stream) {
  (void)in_sizes; (void)n_in; (void)out_size; (void)ws_size;
  const float* x        = (const float*)d_in[0];
  // d_in[1] = idx_base (unused; recomputed implicitly)
  const float* att_w1   = (const float*)d_in[2];
  const float* att_w2   = (const float*)d_in[3];
  const float* upd_w    = (const float*)d_in[4];
  const float* bn_gamma = (const float*)d_in[5];
  const float* bn_beta  = (const float*)d_in[6];
  const float* res_w    = (const float*)d_in[7];
  float* out = (float*)d_out;

  int*   idx_ws  = (int*)d_ws;                         // BNKTOT ints
  float* dist_ws = (float*)d_ws + BNKTOT;              // BNKTOT floats
  float* att_ws  = (float*)d_ws + 2 * (size_t)BNKTOT;  // BNKTOT floats
  float* stats   = (float*)d_ws + 3 * (size_t)BNKTOT;  // 128 floats

  dim3 grid(NN / 16, BB);
  size_t lds1 = 131072 + 8192 + 131072;  // 270336 B
  size_t lds2 = 384 * 96 * 2 + 64 * 96 * 2 + 64 * 32 * 2 + 384 * 32 * 2
              + (384 + 384 + 128) * 4 + 16 * 32 * 2;   // 119296 B

  topk_kernel<<<grid, 256, lds1, stream>>>(x, idx_ws, dist_ws);
  zero_stats_kernel<<<1, 128, 0, stream>>>(stats);
  attn_kernel<<<grid, 256, lds2, stream>>>(x, att_w1, att_w2, upd_w,
                                           idx_ws, dist_ws, att_ws, stats);
  finalize_kernel<<<grid, 256, 0, stream>>>(x, upd_w, res_w, bn_gamma, bn_beta,
                                            idx_ws, att_ws, stats, out);
}
